// LightLIF_29300266893640
// MI455X (gfx1250) — matrix operand order repaired and checked by reading, not compiled, not run on hardware
//
#include <hip/hip_runtime.h>

// ---------------------------------------------------------------------------
// LIF step on MI455X (gfx1250).
// Fast path (needs 37.75 MB of d_ws):
//   K1: convert [inputs|z] f32 -> bf16, tile-packed              (once/elem)
//   K2: convert [w_in;w_rec*mask] f32 -> bf16, transposed+packed (once/elem)
//   K3: bf16 WMMA GEMM (f32 accum), double-buffered LDS (1 barrier/tile),
//       fused LIF epilogue.
// Fallback (small ws): fused convert-in-loop kernel.
// ---------------------------------------------------------------------------

#define N_IN   1024
#define N_REC  2048
#define BATCH  4096
#define K_TOT  (N_IN + N_REC)        // 3072

#define BM     128
#define BN     128
#define BK     32
#define KT     (K_TOT / BK)          // 96 K-tiles
#define CHUNK  (128 * 32)            // elements per packed tile (8 KB bf16)
#define LSTRIDE 40                   // padded LDS stride (bf16 units)

typedef __bf16 v16bf __attribute__((ext_vector_type(16)));
typedef float  v8f   __attribute__((ext_vector_type(8)));
typedef unsigned long long u64;

union Frag {
    u64 q[4];
    v16bf v;
};

__device__ __forceinline__ unsigned short f2bf(float f) {
    unsigned int u = __builtin_bit_cast(unsigned int, f);
    u += 0x7FFFu + ((u >> 16) & 1u);          // round-to-nearest-even
    return (unsigned short)(u >> 16);
}

__device__ __forceinline__ u64 pack4(float a, float b, float c, float d) {
    return (u64)f2bf(a) | ((u64)f2bf(b) << 16) | ((u64)f2bf(c) << 32) | ((u64)f2bf(d) << 48);
}

// ---------------------------------------------------------------------------
// K1: A = [inputs | z] -> bf16, packed [mBlk][kTile][row 128][k 32]
// ---------------------------------------------------------------------------
__global__ __launch_bounds__(256) void cvt_A_kernel(
    const float* __restrict__ inputs, const float* __restrict__ zin,
    unsigned short* __restrict__ Abf)
{
    const int kt   = blockIdx.x;            // 0..95
    const int mBlk = blockIdx.y;            // 0..31
    const int t    = threadIdx.x;

    const bool  ph1  = (kt < N_IN / BK);
    const float* src = ph1 ? inputs : zin;
    const int   srcK = ph1 ? N_IN : N_REC;
    const int   k0   = ph1 ? kt * BK : (kt - N_IN / BK) * BK;

    unsigned short* dst = Abf + ((size_t)mBlk * KT + kt) * CHUNK;
    const int c4 = (t & 7) * 4;
    #pragma unroll
    for (int p = 0; p < 4; ++p) {
        const int r = p * 32 + (t >> 3);
        const float4 f = *(const float4*)(src + (size_t)(mBlk * BM + r) * srcK + k0 + c4);
        *(u64*)(dst + r * BK + c4) = pack4(f.x, f.y, f.z, f.w);
    }
}

// ---------------------------------------------------------------------------
// K2: B = [w_in ; w_rec*(1-I)] -> bf16, transposed+packed
//     [nBlk][kTile][col 128][k 32]   (LDS transpose for coalescing)
// ---------------------------------------------------------------------------
__global__ __launch_bounds__(256) void cvt_B_kernel(
    const float* __restrict__ w_in, const float* __restrict__ w_rec,
    unsigned short* __restrict__ Bbf)
{
    __shared__ __align__(16) unsigned short lt[128 * LSTRIDE];

    const int kt   = blockIdx.x;            // 0..95
    const int nBlk = blockIdx.y;            // 0..15
    const int t    = threadIdx.x;

    const bool   ph1 = (kt < N_IN / BK);
    const float* W   = ph1 ? w_in : w_rec;
    const int    k0  = ph1 ? kt * BK : (kt - N_IN / BK) * BK;
    const int nBase  = nBlk * BN;

    const int c4 = (t & 7) * 4;
    #pragma unroll
    for (int p = 0; p < 4; ++p) {
        const int r    = p * 32 + (t >> 3);
        const int wrow = k0 + r;
        const float4 f = *(const float4*)(W + (size_t)wrow * N_REC + nBase + c4);
        float fv[4] = {f.x, f.y, f.z, f.w};
        #pragma unroll
        for (int i = 0; i < 4; ++i) {
            float val = fv[i];
            if (!ph1 && wrow == nBase + c4 + i) val = 0.0f;   // zero self-recurrence
            lt[(c4 + i) * LSTRIDE + r] = f2bf(val);
        }
    }
    __syncthreads();

    unsigned short* dst = Bbf + ((size_t)nBlk * KT + kt) * CHUNK;
    const int col = t >> 1, half = t & 1;
    const u64* s = (const u64*)&lt[col * LSTRIDE + half * 16];
    u64* d = (u64*)(dst + t * 16);
    d[0] = s[0]; d[1] = s[1]; d[2] = s[2]; d[3] = s[3];
}

// ---------------------------------------------------------------------------
// K3: bf16 WMMA GEMM + LIF epilogue. Double-buffered LDS, 1 barrier/tile.
// ---------------------------------------------------------------------------
__global__ __launch_bounds__(256) void lif_gemm_bf16_kernel(
    const unsigned short* __restrict__ Abf,
    const unsigned short* __restrict__ Bbf,
    const float* __restrict__ vmem, const float* __restrict__ zin,
    float* __restrict__ out)
{
    const float DECAY   = 0.9512294245f;
    const float ONE_MD  = 0.0487705755f;
    const float THR     = 0.615f;
    const float INV_THR = 1.0f / 0.615f;

    __shared__ __align__(16) unsigned short lA[2][128 * LSTRIDE];
    __shared__ __align__(16) unsigned short lB[2][128 * LSTRIDE];

    const int tid   = threadIdx.x;
    const int lane  = tid & 31;
    const int wave  = tid >> 5;
    const int waveM = wave >> 2;
    const int waveN = wave & 3;
    const int nBlk  = blockIdx.x, mBlk = blockIdx.y;

    const unsigned short* aChunk = Abf + (size_t)mBlk * KT * CHUNK;
    const unsigned short* bChunk = Bbf + (size_t)nBlk * KT * CHUNK;

    const int srow  = tid >> 1;       // staging: 32 B per thread
    const int shalf = tid & 1;
    const int lrow  = lane & 15;
    const int kbA   = (lane >> 4) * 16;
    const int kloB  = (lane >> 4) * 16;

    v8f acc[4][2];
    #pragma unroll
    for (int m = 0; m < 4; ++m)
        #pragma unroll
        for (int n = 0; n < 2; ++n)
            acc[m][n] = (v8f)0.0f;

    uint4 a0, a1, b0, b1;
    auto loadChunk = [&](int kt) {
        const uint4* ga = (const uint4*)(aChunk + (size_t)kt * CHUNK);
        const uint4* gb = (const uint4*)(bChunk + (size_t)kt * CHUNK);
        a0 = ga[tid * 2]; a1 = ga[tid * 2 + 1];
        b0 = gb[tid * 2]; b1 = gb[tid * 2 + 1];
        if (kt + 1 < KT) {            // keep next chunks warm in near cache
            __builtin_prefetch(aChunk + (size_t)(kt + 1) * CHUNK + tid * 16, 0, 1);
            __builtin_prefetch(bChunk + (size_t)(kt + 1) * CHUNK + tid * 16, 0, 1);
        }
    };
    auto storeChunk = [&](int buf) {
        *(uint4*)(&lA[buf][srow * LSTRIDE + shalf * 16])     = a0;
        *(uint4*)(&lA[buf][srow * LSTRIDE + shalf * 16 + 8]) = a1;
        *(uint4*)(&lB[buf][srow * LSTRIDE + shalf * 16])     = b0;
        *(uint4*)(&lB[buf][srow * LSTRIDE + shalf * 16 + 8]) = b1;
    };
    auto compute = [&](int buf) {
        Frag am[4], bn[2];
        #pragma unroll
        for (int m = 0; m < 4; ++m) {
            const int r = waveM * 64 + m * 16 + lrow;
            const u64* p0 = (const u64*)&lA[buf][r * LSTRIDE + kbA];
            am[m].q[0] = p0[0]; am[m].q[1] = p0[1];
            const u64* p1 = (const u64*)&lA[buf][r * LSTRIDE + kbA + 8];
            am[m].q[2] = p1[0]; am[m].q[3] = p1[1];
        }
        #pragma unroll
        for (int n = 0; n < 2; ++n) {
            const int c = waveN * 32 + n * 16 + lrow;
            const u64* p = (const u64*)&lB[buf][c * LSTRIDE + kloB];
            bn[n].q[0] = p[0]; bn[n].q[1] = p[1];
            bn[n].q[2] = p[2]; bn[n].q[3] = p[3];
        }
        #pragma unroll
        for (int m = 0; m < 4; ++m)
            #pragma unroll
            for (int n = 0; n < 2; ++n)
                acc[m][n] = __builtin_amdgcn_wmma_f32_16x16x32_bf16(
                    false, am[m].v, false, bn[n].v,
                    (short)0, acc[m][n], false, false);
    };

    // prologue: stage tile 0 into buffer 0
    loadChunk(0);
    storeChunk(0);
    __syncthreads();

    for (int kt = 0; kt < KT; kt += 2) {          // KT is even
        loadChunk(kt + 1);        // global loads in flight during compute
        compute(0);
        storeChunk(1);            // waits loadcnt, not other waves
        __syncthreads();          // buf1 visible; buf0 reads done

        if (kt + 2 < KT) loadChunk(kt + 2);
        compute(1);
        if (kt + 2 < KT) storeChunk(0);
        __syncthreads();          // buf0 visible; buf1 reads done
    }

    // ---- LIF epilogue
    const size_t outV = (size_t)BATCH * N_REC;
    #pragma unroll
    for (int m = 0; m < 4; ++m) {
        #pragma unroll
        for (int n = 0; n < 2; ++n) {
            const int r0 = mBlk * BM + waveM * 64 + m * 16 + ((lane >> 4) * 8);
            const int c  = nBlk * BN + waveN * 32 + n * 16 + lrow;
            #pragma unroll
            for (int j = 0; j < 8; ++j) {
                const size_t idx = (size_t)(r0 + j) * N_REC + c;
                const float nv = fmaf(DECAY, vmem[idx],
                                  fmaf(ONE_MD, acc[m][n][j], -zin[idx] * THR));
                const float vs = (nv - THR) * INV_THR;
                out[idx]        = (vs > THR) ? 1.0f : 0.0f;
                out[idx + outV] = nv;
            }
        }
    }
}

// ---------------------------------------------------------------------------
// Fallback: fused kernel (convert inside the K loop), for small d_ws.
// ---------------------------------------------------------------------------
__global__ __launch_bounds__(256) void lif_fused_kernel(
    const float* __restrict__ inputs, const float* __restrict__ vmem,
    const float* __restrict__ zin, const float* __restrict__ w_in,
    const float* __restrict__ w_rec, float* __restrict__ out)
{
    const float DECAY = 0.9512294245f, ONE_MD = 0.0487705755f;
    const float THR = 0.615f, INV_THR = 1.0f / 0.615f;

    __shared__ __align__(16) unsigned short lA[BM * LSTRIDE];
    __shared__ __align__(16) unsigned short lB[BN * LSTRIDE];

    const int tid = threadIdx.x, lane = tid & 31, wave = tid >> 5;
    const int waveM = wave >> 2, waveN = wave & 3;
    const int nBase = blockIdx.x * BN, mBase = blockIdx.y * BM;
    const int lrow = lane & 15, kbA = (lane >> 4) * 16, kloB = (lane >> 4) * 16;

    v8f acc[4][2];
    #pragma unroll
    for (int m = 0; m < 4; ++m)
        #pragma unroll
        for (int n = 0; n < 2; ++n) acc[m][n] = (v8f)0.0f;

    const int bCol = tid & 127, bKg = (tid >> 7) * 16, gn = nBase + bCol;

    for (int k0 = 0; k0 < K_TOT; k0 += BK) {
        const bool ph1 = (k0 < N_IN);
        const float* Ap = ph1 ? inputs : zin;
        const int aK = ph1 ? N_IN : N_REC, ak0 = ph1 ? k0 : (k0 - N_IN);
        const float* Wp = ph1 ? w_in : w_rec;

        #pragma unroll
        for (int p = 0; p < 4; ++p) {
            const int row = p * 32 + (tid >> 3), col = (tid & 7) * 4;
            const float4 f = *(const float4*)(Ap + (size_t)(mBase + row) * aK + ak0 + col);
            *(u64*)(&lA[row * LSTRIDE + col]) = pack4(f.x, f.y, f.z, f.w);
        }
        #pragma unroll
        for (int g = 0; g < 4; ++g) {
            float f[4];
            #pragma unroll
            for (int i = 0; i < 4; ++i) {
                const int wrow = ak0 + bKg + g * 4 + i;
                float val = Wp[(size_t)wrow * N_REC + gn];
                if (!ph1 && wrow == gn) val = 0.0f;
                f[i] = val;
            }
            *(u64*)(&lB[bCol * LSTRIDE + bKg + g * 4]) = pack4(f[0], f[1], f[2], f[3]);
        }
        __syncthreads();

        Frag am[4], bn[2];
        #pragma unroll
        for (int m = 0; m < 4; ++m) {
            const int r = waveM * 64 + m * 16 + lrow;
            const u64* p0 = (const u64*)&lA[r * LSTRIDE + kbA];
            am[m].q[0] = p0[0]; am[m].q[1] = p0[1];
            const u64* p1 = (const u64*)&lA[r * LSTRIDE + kbA + 8];
            am[m].q[2] = p1[0]; am[m].q[3] = p1[1];
        }
        #pragma unroll
        for (int n = 0; n < 2; ++n) {
            const int c = waveN * 32 + n * 16 + lrow;
            const u64* p = (const u64*)&lB[c * LSTRIDE + kloB];
            bn[n].q[0] = p[0]; bn[n].q[1] = p[1]; bn[n].q[2] = p[2]; bn[n].q[3] = p[3];
        }
        #pragma unroll
        for (int m = 0; m < 4; ++m)
            #pragma unroll
            for (int n = 0; n < 2; ++n)
                acc[m][n] = __builtin_amdgcn_wmma_f32_16x16x32_bf16(
                    false, am[m].v, false, bn[n].v, (short)0, acc[m][n], false, false);
        __syncthreads();
    }

    const size_t outV = (size_t)BATCH * N_REC;
    #pragma unroll
    for (int m = 0; m < 4; ++m)
        #pragma unroll
        for (int n = 0; n < 2; ++n) {
            const int r0 = mBase + waveM * 64 + m * 16 + ((lane >> 4) * 8);
            const int c  = nBase + waveN * 32 + n * 16 + lrow;
            #pragma unroll
            for (int j = 0; j < 8; ++j) {
                const size_t idx = (size_t)(r0 + j) * N_REC + c;
                const float nv = fmaf(DECAY, vmem[idx],
                                  fmaf(ONE_MD, acc[m][n][j], -zin[idx] * THR));
                const float vs = (nv - THR) * INV_THR;
                out[idx]        = (vs > THR) ? 1.0f : 0.0f;
                out[idx + outV] = nv;
            }
        }
}

extern "C" void kernel_launch(void* const* d_in, const int* in_sizes, int n_in,
                              void* d_out, int out_size, void* d_ws, size_t ws_size,
                              hipStream_t stream) {
    const float* inputs = (const float*)d_in[0];
    const float* v      = (const float*)d_in[1];
    const float* z      = (const float*)d_in[2];
    const float* w_in   = (const float*)d_in[3];
    const float* w_rec  = (const float*)d_in[4];
    float* out          = (float*)d_out;

    const size_t aBytes = (size_t)BATCH * K_TOT * 2;   // 25,165,824
    const size_t bBytes = (size_t)K_TOT * N_REC * 2;   // 12,582,912

    if (ws_size >= aBytes + bBytes) {
        unsigned short* Abf = (unsigned short*)d_ws;
        unsigned short* Bbf = (unsigned short*)((char*)d_ws + aBytes);
        cvt_A_kernel<<<dim3(KT, BATCH / BM), 256, 0, stream>>>(inputs, z, Abf);
        cvt_B_kernel<<<dim3(KT, N_REC / BN), 256, 0, stream>>>(w_in, w_rec, Bbf);
        lif_gemm_bf16_kernel<<<dim3(N_REC / BN, BATCH / BM), 256, 0, stream>>>(
            Abf, Bbf, v, z, out);
    } else {
        lif_fused_kernel<<<dim3(N_REC / BN, BATCH / BM), 256, 0, stream>>>(
            inputs, v, z, w_in, w_rec, out);
    }
}
